// Attention_70660801953920
// MI455X (gfx1250) — matrix operand order repaired and checked by reading, not compiled, not run on hardware
//
#include <hip/hip_runtime.h>

// Shapes from the reference
#define S_DIM 128
#define R_DIM 256
#define C_DIM 256
#define H_DIM 8
#define CA    32
#define NORM  0.17677669529663687f  // 1/sqrt(32)

typedef __attribute__((ext_vector_type(16))) _Float16     v16h;
typedef __attribute__((ext_vector_type(4)))  _Float16     v4h;
typedef __attribute__((ext_vector_type(8)))  float        v8f;
typedef __attribute__((ext_vector_type(4)))  float        v4f;
typedef __attribute__((ext_vector_type(4)))  unsigned int v4u;

union U16H { v16h v; unsigned int u[8]; v4u q[2]; };
union H2   { _Float16 h[2]; unsigned int u; };
struct __align__(8) F2 { float x, y; };

__device__ inline unsigned int pack2h(float a, float b) {
  H2 t; t.h[0] = (_Float16)a; t.h[1] = (_Float16)b; return t.u;
}

__device__ inline v8f wmma_f16(v16h a, v16h b, v8f c) {
  return __builtin_amdgcn_wmma_f32_16x16x32_f16(false, a, false, b, (short)0, c,
                                                false, false);
}

#define V8F_ZERO {0.f, 0.f, 0.f, 0.f, 0.f, 0.f, 0.f, 0.f}

// ---------------------------------------------------------------------------
// Fragment builders (contiguous k = 16*hi + s per lane half, wave32)
// ---------------------------------------------------------------------------

// A fragment (16x32 f16) from LDS half tile [16][pitch] @ col kbase.
// Two contiguous 16B runs per lane -> two ds_load_b128.
__device__ inline v16h a_frag_lds(const _Float16* tile, int pitch, int kbase,
                                  int m, int hi) {
  U16H r;
  const _Float16* p = tile + m * pitch + kbase + hi * 16;
  r.q[0] = *(const v4u*)(p);
  r.q[1] = *(const v4u*)(p + 8);
  return r.v;
}

// A fragment from LDS fp32 score rows [16][256], scaled by softmax 1/sum.
__device__ inline v16h a_frag_scores(const float* sc, int kbase, float scale,
                                     int m, int hi) {
  U16H r;
#pragma unroll
  for (int v = 0; v < 8; ++v) {
    int k0 = (hi << 4) + (v << 1) + kbase;
    F2 f = *(const F2*)(sc + m * 256 + k0);
    r.u[v] = pack2h(f.x * scale, f.y * scale);
  }
  return r.v;
}

// B fragment from pre-packed weights Wp[kt2][256][8] u32 -> two b128 loads.
__device__ inline v16h b_frag_wp(const unsigned int* __restrict__ Wp, int kt,
                                 int cb, int n, int hi) {
  U16H r;
  const unsigned int* p = Wp + ((((kt << 1) + hi) << 8) + cb + n) * 8;
  r.q[0] = *(const v4u*)(p);
  r.q[1] = *(const v4u*)(p + 4);
  return r.v;
}

// B fragment for Q*K^T: B[d][kr] = K[kr][dbase+d]; K f16 row-major [R][256].
__device__ inline v16h b_frag_kT(const _Float16* __restrict__ Kp, int kr,
                                 int dbase, int hi) {
  U16H r;
  const _Float16* p = Kp + (size_t)kr * 256 + dbase + hi * 16;
  r.q[0] = *(const v4u*)(p);
  r.q[1] = *(const v4u*)(p + 8);
  return r.v;
}

// B fragment for P*V with V stored TRANSPOSED: Vt[hd][kr] f16.
__device__ inline v16h b_frag_vT(const _Float16* __restrict__ Vt, int kbase,
                                 int col, int hi) {
  U16H r;
  const _Float16* p = Vt + (size_t)col * 256 + kbase + hi * 16;
  r.q[0] = *(const v4u*)(p);
  r.q[1] = *(const v4u*)(p + 8);
  return r.v;
}

// Stage a 16x256 fp32 tile into LDS f16 [16][264] with b128 loads.
__device__ inline void stage_tile_f16(const float* __restrict__ src,
                                      _Float16* dst) {
  for (int i = threadIdx.x; i < 1024; i += 256) {   // 4 quads per thread
    int r = i >> 6, c4 = (i & 63) << 2;
    v4f f = *(const v4f*)(src + r * 256 + c4);
    v4h hx;
    hx[0] = (_Float16)f[0]; hx[1] = (_Float16)f[1];
    hx[2] = (_Float16)f[2]; hx[3] = (_Float16)f[3];
    *(v4h*)(dst + r * 264 + c4) = hx;               // 8B aligned (528B pitch)
  }
}

// ---------------------------------------------------------------------------
// Kernel 0: pack one 256x256 fp32 weight into B-fragment layout.
// ---------------------------------------------------------------------------
__global__ __launch_bounds__(256) void pack_w_kernel(
    const float* __restrict__ W, unsigned int* __restrict__ Wp) {
  int t = blockIdx.x * 256 + threadIdx.x;   // 0 .. 32767
  int v = t & 7, c = (t >> 3) & 255, kt2 = t >> 11;
  int k = ((kt2 >> 1) << 5) + ((kt2 & 1) << 4) + (v << 1);
  Wp[t] = pack2h(W[(size_t)k * 256 + c], W[(size_t)(k + 1) * 256 + c]);
}

// ---------------------------------------------------------------------------
// Kernel 1: K = kv@w_k (row-major f16), V = kv@w_v (transposed f16).
// ---------------------------------------------------------------------------
__global__ __launch_bounds__(256) void proj_kv_kernel(
    const float* __restrict__ kv, const unsigned int* __restrict__ wkp,
    const unsigned int* __restrict__ wvp, _Float16* __restrict__ Kw,
    _Float16* __restrict__ Vt) {
  const int s = blockIdx.x, rt = blockIdx.y;
  __shared__ _Float16 st[16 * 264];

  stage_tile_f16(kv + ((size_t)s * R_DIM + rt * 16) * C_DIM, st);
  __syncthreads();

  const int wave = threadIdx.x >> 5;
  const int lane = threadIdx.x & 31;
  const int n = lane & 15, hi = lane >> 4, m = lane & 15;

  for (int task = wave; task < 16; task += 8) {
    const unsigned int* Wp = (task < 8) ? wkp : wvp;
    const int cb = (task & 7) * 32;
    v8f acc0 = V8F_ZERO, acc1 = V8F_ZERO;
    // software-pipelined: fetch kk+1 fragments before kk's WMMAs
    v16h a  = a_frag_lds(st, 264, 0, m, hi);
    v16h b0 = b_frag_wp(Wp, 0, cb, n, hi);
    v16h b1 = b_frag_wp(Wp, 0, cb + 16, n, hi);
#pragma unroll
    for (int kk = 0; kk < 8; ++kk) {
      v16h an = a, bn0 = b0, bn1 = b1;
      if (kk < 7) {
        an  = a_frag_lds(st, 264, (kk + 1) * 32, m, hi);
        bn0 = b_frag_wp(Wp, kk + 1, cb, n, hi);
        bn1 = b_frag_wp(Wp, kk + 1, cb + 16, n, hi);
      }
      acc0 = wmma_f16(a, b0, acc0);
      acc1 = wmma_f16(a, b1, acc1);
      a = an; b0 = bn0; b1 = bn1;
    }
    if (task < 8) {  // K: row-major [r][hd]
#pragma unroll
      for (int v = 0; v < 8; ++v) {
        int row = v + hi * 8;
        size_t rbase = ((size_t)s * R_DIM + rt * 16 + row) * 256;
        Kw[rbase + cb + n]      = (_Float16)acc0[v];
        Kw[rbase + cb + 16 + n] = (_Float16)acc1[v];
      }
    } else {  // V: transposed [hd][r]
#pragma unroll
      for (int v = 0; v < 8; ++v) {
        int row = v + hi * 8;
        Vt[((size_t)s * 256 + cb + n) * 256 + rt * 16 + row]      = (_Float16)acc0[v];
        Vt[((size_t)s * 256 + cb + 16 + n) * 256 + rt * 16 + row] = (_Float16)acc1[v];
      }
    }
  }
}

// ---------------------------------------------------------------------------
// Kernel 2: fused per (s, q-tile): Q proj, QK^T, softmax(+bias+mask), P@V,
// sigmoid gate, output proj. grid (S, R/16), 256 threads; wave == head.
// ---------------------------------------------------------------------------
__global__ __launch_bounds__(256) void attn_kernel(
    const float* __restrict__ q, const float* __restrict__ bias,
    const float* __restrict__ bmask, const unsigned int* __restrict__ wqp,
    const unsigned int* __restrict__ wgp, const float* __restrict__ b_g,
    const unsigned int* __restrict__ wop, const float* __restrict__ b_o,
    const _Float16* __restrict__ Kw, const _Float16* __restrict__ Vt,
    float* __restrict__ out) {
  const int s = blockIdx.x, qt = blockIdx.y;

  __shared__ _Float16 qstage[16 * 264];     // q tile f16; reused for gated O
  __shared__ _Float16 qh[H_DIM][16 * 40];   // per-head Q_h f16 (16x32)
  __shared__ float    sc[H_DIM][16 * 256];  // per-head score rows (fp32)

  stage_tile_f16(q + ((size_t)s * R_DIM + qt * 16) * C_DIM, qstage);
  __syncthreads();

  const int h = threadIdx.x >> 5;  // wave == head
  const int lane = threadIdx.x & 31;
  const int n = lane & 15, hi = lane >> 4, m = lane & 15;
  const _Float16* Kp  = Kw + (size_t)s * R_DIM * 256;
  const _Float16* Vtp = Vt + (size_t)s * 256 * R_DIM;

  // ---- Q projection (16x32), software-pipelined ----
  {
    v8f acc0 = V8F_ZERO, acc1 = V8F_ZERO;
    v16h a  = a_frag_lds(qstage, 264, 0, m, hi);
    v16h b0 = b_frag_wp(wqp, 0, h * 32, n, hi);
    v16h b1 = b_frag_wp(wqp, 0, h * 32 + 16, n, hi);
#pragma unroll
    for (int kk = 0; kk < 8; ++kk) {
      v16h an = a, bn0 = b0, bn1 = b1;
      if (kk < 7) {
        an  = a_frag_lds(qstage, 264, (kk + 1) * 32, m, hi);
        bn0 = b_frag_wp(wqp, kk + 1, h * 32, n, hi);
        bn1 = b_frag_wp(wqp, kk + 1, h * 32 + 16, n, hi);
      }
      acc0 = wmma_f16(a, b0, acc0);
      acc1 = wmma_f16(a, b1, acc1);
      a = an; b0 = bn0; b1 = bn1;
    }
#pragma unroll
    for (int v = 0; v < 8; ++v) {
      int row = v + hi * 8;
      qh[h][row * 40 + n]      = (_Float16)(acc0[v] * NORM);
      qh[h][row * 40 + 16 + n] = (_Float16)(acc1[v] * NORM);
    }
  }

  // ---- QK^T: scores[16q][256k] (A invariant; B double-buffered) ----
  {
    v16h a = a_frag_lds(qh[h], 40, 0, m, hi);
    v16h b = b_frag_kT(Kp, n, h * 32, hi);
    for (int kt = 0; kt < 16; ++kt) {
      v16h bn = b;
      if (kt < 15) bn = b_frag_kT(Kp, (kt + 1) * 16 + n, h * 32, hi);
      v8f acc = V8F_ZERO;
      acc = wmma_f16(a, b, acc);
#pragma unroll
      for (int v = 0; v < 8; ++v)
        sc[h][(v + hi * 8) * 256 + kt * 16 + n] = acc[v];
      b = bn;
    }
  }

  // ---- bias + mask + softmax over k (lane pair (l, l^16) shares a row) ----
  float rinv;
  {
    const int row = n;
    float* srow = &sc[h][row * 256 + hi * 128];
    const int qg = qt * 16 + row;
    const float* brow = bias + ((size_t)qg * 256 + hi * 128) * H_DIM + h;
    const float* bmr = bmask + (size_t)s * 256 + hi * 128;
    float mx = -1e30f;
    for (int k = 0; k < 128; ++k) {
      float t = srow[k] + brow[k * 8] + (bmr[k] - 1.0f) * 1e9f;
      srow[k] = t;
      mx = fmaxf(mx, t);
    }
    mx = fmaxf(mx, __shfl_xor(mx, 16, 32));
    float sum = 0.f;
    for (int k = 0; k < 128; ++k) {
      float e = __expf(srow[k] - mx);
      srow[k] = e;
      sum += e;
    }
    sum += __shfl_xor(sum, 16, 32);
    rinv = 1.0f / sum;
  }

  // ---- O = P @ V (16x32), V transposed; B double-buffered ----
  v8f o0 = V8F_ZERO, o1 = V8F_ZERO;
  {
    v16h a  = a_frag_scores(sc[h], 0, rinv, m, hi);
    v16h b0 = b_frag_vT(Vtp, 0, h * 32 + n, hi);
    v16h b1 = b_frag_vT(Vtp, 0, h * 32 + 16 + n, hi);
#pragma unroll
    for (int kk = 0; kk < 8; ++kk) {
      v16h an = a, bn0 = b0, bn1 = b1;
      if (kk < 7) {
        an  = a_frag_scores(sc[h], (kk + 1) * 32, rinv, m, hi);
        bn0 = b_frag_vT(Vtp, (kk + 1) * 32, h * 32 + n, hi);
        bn1 = b_frag_vT(Vtp, (kk + 1) * 32, h * 32 + 16 + n, hi);
      }
      o0 = wmma_f16(a, b0, o0);
      o1 = wmma_f16(a, b1, o1);
      a = an; b0 = bn0; b1 = bn1;
    }
  }

  // ---- Gate: G = sigmoid(q@w_g + b_g); O *= G (matching D layouts) ----
  {
    v8f g0 = V8F_ZERO, g1 = V8F_ZERO;
    v16h a  = a_frag_lds(qstage, 264, 0, m, hi);
    v16h b0 = b_frag_wp(wgp, 0, h * 32, n, hi);
    v16h b1 = b_frag_wp(wgp, 0, h * 32 + 16, n, hi);
#pragma unroll
    for (int kk = 0; kk < 8; ++kk) {
      v16h an = a, bn0 = b0, bn1 = b1;
      if (kk < 7) {
        an  = a_frag_lds(qstage, 264, (kk + 1) * 32, m, hi);
        bn0 = b_frag_wp(wgp, kk + 1, h * 32, n, hi);
        bn1 = b_frag_wp(wgp, kk + 1, h * 32 + 16, n, hi);
      }
      g0 = wmma_f16(a, b0, g0);
      g1 = wmma_f16(a, b1, g1);
      a = an; b0 = bn0; b1 = bn1;
    }
    const float bg0 = b_g[h * 32 + n], bg1 = b_g[h * 32 + 16 + n];
#pragma unroll
    for (int v = 0; v < 8; ++v) {
      o0[v] *= 1.0f / (1.0f + __expf(-(g0[v] + bg0)));
      o1[v] *= 1.0f / (1.0f + __expf(-(g1[v] + bg1)));
    }
  }

  // ---- Stage gated O (f16) into qstage (reuse) ----
  __syncthreads();  // all waves done reading q tile
#pragma unroll
  for (int v = 0; v < 8; ++v) {
    int row = v + hi * 8;
    qstage[row * 264 + h * 32 + n]      = (_Float16)o0[v];
    qstage[row * 264 + h * 32 + 16 + n] = (_Float16)o1[v];
  }
  __syncthreads();

  // ---- Output projection: wave h computes cols [h*32, h*32+32) ----
  {
    v8f acc0 = V8F_ZERO, acc1 = V8F_ZERO;
    v16h a  = a_frag_lds(qstage, 264, 0, m, hi);
    v16h b0 = b_frag_wp(wop, 0, h * 32, n, hi);
    v16h b1 = b_frag_wp(wop, 0, h * 32 + 16, n, hi);
#pragma unroll
    for (int kk = 0; kk < 8; ++kk) {
      v16h an = a, bn0 = b0, bn1 = b1;
      if (kk < 7) {
        an  = a_frag_lds(qstage, 264, (kk + 1) * 32, m, hi);
        bn0 = b_frag_wp(wop, kk + 1, h * 32, n, hi);
        bn1 = b_frag_wp(wop, kk + 1, h * 32 + 16, n, hi);
      }
      acc0 = wmma_f16(a, b0, acc0);
      acc1 = wmma_f16(a, b1, acc1);
      a = an; b0 = bn0; b1 = bn1;
    }
    const int c0 = h * 32 + n, c1 = h * 32 + 16 + n;
    const float bo0 = b_o[c0], bo1 = b_o[c1];
#pragma unroll
    for (int v = 0; v < 8; ++v) {
      size_t rbase = ((size_t)s * R_DIM + qt * 16 + v + hi * 8) * C_DIM;
      out[rbase + c0] = acc0[v] + bo0;
      out[rbase + c1] = acc1[v] + bo1;
    }
  }
}

// ---------------------------------------------------------------------------
extern "C" void kernel_launch(void* const* d_in, const int* in_sizes, int n_in,
                              void* d_out, int out_size, void* d_ws,
                              size_t ws_size, hipStream_t stream) {
  const float* q     = (const float*)d_in[0];
  const float* kv    = (const float*)d_in[1];
  const float* bias  = (const float*)d_in[2];
  const float* bmask = (const float*)d_in[3];
  const float* w_q   = (const float*)d_in[4];
  const float* w_k   = (const float*)d_in[5];
  const float* w_v   = (const float*)d_in[6];
  const float* w_g   = (const float*)d_in[7];
  const float* b_g   = (const float*)d_in[8];
  const float* w_o   = (const float*)d_in[9];
  const float* b_o   = (const float*)d_in[10];
  float* out = (float*)d_out;

  // Workspace: 5 packed weights | Kw f16 [S][R][256] | Vt f16 [S][256][R]
  const size_t WPK = 16 * 256 * 8;  // 32768 u32 per packed weight
  unsigned int* wqp = (unsigned int*)d_ws;
  unsigned int* wkp = wqp + WPK;
  unsigned int* wvp = wkp + WPK;
  unsigned int* wgp = wvp + WPK;
  unsigned int* wop = wgp + WPK;
  const size_t kv_elems = (size_t)S_DIM * R_DIM * 256;
  _Float16* Kw = (_Float16*)(wop + WPK);
  _Float16* Vt = Kw + kv_elems;

  pack_w_kernel<<<128, 256, 0, stream>>>(w_q, wqp);
  pack_w_kernel<<<128, 256, 0, stream>>>(w_k, wkp);
  pack_w_kernel<<<128, 256, 0, stream>>>(w_v, wvp);
  pack_w_kernel<<<128, 256, 0, stream>>>(w_g, wgp);
  pack_w_kernel<<<128, 256, 0, stream>>>(w_o, wop);

  proj_kv_kernel<<<dim3(S_DIM, R_DIM / 16), 256, 0, stream>>>(kv, wkp, wvp, Kw, Vt);
  attn_kernel<<<dim3(S_DIM, R_DIM / 16), 256, 0, stream>>>(
      q, bias, bmask, wqp, wgp, b_g, wop, b_o, Kw, Vt, out);
}
